// SparseFFN_36326833390147
// MI455X (gfx1250) — matrix operand order repaired and checked by reading, not compiled, not run on hardware
//
#include <hip/hip_runtime.h>
#include <hip/hip_bf16.h>
#include <stdint.h>

#define NE   8
#define KTOP 2
#define CAP  1280
#define DM   1024
#define FF   4096
#define NTOK 4096

typedef __attribute__((ext_vector_type(16))) __bf16 v16bf;
typedef __attribute__((ext_vector_type(8)))  __bf16 v8bf;
typedef __attribute__((ext_vector_type(8)))  float  v8f;

// async-to-LDS builtin takes pointers to 16B int vectors in AS1/AS3
typedef int v4i_gcc __attribute__((vector_size(16)));
typedef __attribute__((address_space(1))) v4i_gcc gv4i;
typedef __attribute__((address_space(3))) v4i_gcc lv4i;

#if defined(__AMDGCN__) && __has_builtin(__builtin_amdgcn_global_load_async_to_lds_b128)
#define HAVE_ASYNC 1
#else
#define HAVE_ASYNC 0
#endif

static __device__ __forceinline__ void async_cp16(const __bf16* g, __bf16* l) {
#if HAVE_ASYNC
  __builtin_amdgcn_global_load_async_to_lds_b128((gv4i*)(void*)const_cast<__bf16*>(g),
                                                 (lv4i*)l, 0, 0);
#else
  *(v8bf*)l = *(const v8bf*)g;
#endif
}
static __device__ __forceinline__ void async_wait() {
#if HAVE_ASYNC
#if __has_builtin(__builtin_amdgcn_s_wait_asynccnt)
  __builtin_amdgcn_s_wait_asynccnt(0);
#else
  asm volatile("s_wait_asynccnt 0x0" ::: "memory");
#endif
#endif
}

static __device__ __forceinline__ __bf16 to_bf16(float f) {
  union { float f; uint32_t u; } in; in.f = f;
  uint32_t r = (in.u + 0x7FFFu + ((in.u >> 16) & 1u)) >> 16;
  union { uint16_t s; __bf16 b; } out; out.s = (uint16_t)r;
  return out.b;
}

// ---------------------------------------------------------------- zero output
__global__ void k_zero(float* __restrict__ p, int n4) {
  int i = blockIdx.x * blockDim.x + threadIdx.x;
  if (i < n4) {
    float4 z = make_float4(0.f, 0.f, 0.f, 0.f);
    *(float4*)(p + (size_t)i * 4) = z;
  }
}

// ---------------------------------------------------------------- router
__global__ void k_router(const float* __restrict__ x, const float* __restrict__ Wr,
                         const float* __restrict__ br,
                         float* __restrict__ logits, float* __restrict__ probs,
                         int* __restrict__ tki, float* __restrict__ tkp) {
  __shared__ float sWr[DM * NE];
  for (int i = threadIdx.x; i < DM * NE; i += blockDim.x) sWr[i] = Wr[i];
  __syncthreads();
  int n = blockIdx.x * blockDim.x + threadIdx.x;
  if (n >= NTOK) return;
  float acc[NE];
#pragma unroll
  for (int e = 0; e < NE; ++e) acc[e] = br[e];
  const float* xr = x + (size_t)n * DM;
  for (int d = 0; d < DM; d += 4) {
    float4 xv = *(const float4*)(xr + d);
#pragma unroll
    for (int e = 0; e < NE; ++e) {
      acc[e] += xv.x * sWr[(d + 0) * NE + e] + xv.y * sWr[(d + 1) * NE + e]
              + xv.z * sWr[(d + 2) * NE + e] + xv.w * sWr[(d + 3) * NE + e];
    }
  }
  float mx = acc[0];
#pragma unroll
  for (int e = 1; e < NE; ++e) mx = fmaxf(mx, acc[e]);
  float p[NE]; float s = 0.f;
#pragma unroll
  for (int e = 0; e < NE; ++e) { p[e] = __expf(acc[e] - mx); s += p[e]; }
  float inv = 1.f / s;
#pragma unroll
  for (int e = 0; e < NE; ++e) {
    p[e] *= inv;
    logits[(size_t)n * NE + e] = acc[e];
    probs[(size_t)n * NE + e]  = p[e];
  }
  int i0 = 0; float v0 = p[0];
#pragma unroll
  for (int e = 1; e < NE; ++e) if (p[e] > v0) { v0 = p[e]; i0 = e; }
  int i1 = -1; float v1 = -1.f;
#pragma unroll
  for (int e = 0; e < NE; ++e) if (e != i0 && p[e] > v1) { v1 = p[e]; i1 = e; }
  tki[n * KTOP + 0] = i0; tkp[n * KTOP + 0] = v0;
  tki[n * KTOP + 1] = i1; tkp[n * KTOP + 1] = v1;
}

// ------------------------------------------------- FCFS capacity scan (1 blk)
#define DTH 256
__global__ void k_dispatch(const int* __restrict__ tki, const float* __restrict__ tkp,
                           int* __restrict__ eidx_o, float* __restrict__ eprob_o,
                           int* __restrict__ eidx_w, float* __restrict__ eprob_w) {
  __shared__ int cnt[DTH][NE];
  int t = threadIdx.x;
  for (int s = t; s < NE * CAP; s += DTH) {
    eidx_o[s] = -1; eidx_w[s] = -1; eprob_o[s] = 0.f; eprob_w[s] = 0.f;
  }
  const int PER = (NTOK * KTOP) / DTH;  // 32 assignments per thread, token order
  int base = t * PER;
  int lc[NE];
#pragma unroll
  for (int e = 0; e < NE; ++e) lc[e] = 0;
  for (int j = 0; j < PER; ++j) {
    int ei = tki[base + j];
#pragma unroll
    for (int e = 0; e < NE; ++e) if (ei == e) lc[e]++;
  }
#pragma unroll
  for (int e = 0; e < NE; ++e) cnt[t][e] = lc[e];
  __syncthreads();
  for (int off = 1; off < DTH; off <<= 1) {
    int v[NE];
#pragma unroll
    for (int e = 0; e < NE; ++e)
      v[e] = cnt[t][e] + ((t >= off) ? cnt[t - off][e] : 0);
    __syncthreads();
#pragma unroll
    for (int e = 0; e < NE; ++e) cnt[t][e] = v[e];
    __syncthreads();
  }
  int pos[NE];
#pragma unroll
  for (int e = 0; e < NE; ++e) pos[e] = cnt[t][e] - lc[e];  // exclusive prefix
  for (int j = 0; j < PER; ++j) {
    int a = base + j;
    int ei = tki[a];
    float pv = tkp[a];
    int pp = 0;
#pragma unroll
    for (int e = 0; e < NE; ++e) if (ei == e) { pp = pos[e]; pos[e] = pp + 1; }
    if (pp < CAP) {
      int tok = a >> 1;  // a / KTOP
      int s = ei * CAP + pp;
      eidx_o[s] = tok; eprob_o[s] = pv;
      eidx_w[s] = tok; eprob_w[s] = pv;
    }
  }
}

// ---------------------------------------------------------------- gather->bf16
__global__ void k_gather(const float* __restrict__ x, const int* __restrict__ eidx,
                         __bf16* __restrict__ xe) {
  int slot = blockIdx.x;
  int tok = eidx[slot];
  __bf16* dst = xe + (size_t)slot * DM;
  if (tok >= 0) {
    const float* src = x + (size_t)tok * DM;
    for (int i = threadIdx.x; i < DM; i += blockDim.x) dst[i] = to_bf16(src[i]);
  } else {
    for (int i = threadIdx.x; i < DM; i += blockDim.x) dst[i] = to_bf16(0.f);
  }
}

// ---------------------------------------------------------------- WMMA GEMMs
#define BM  128
#define BN  128
#define BK  32
#define LDA 40
#define LDB 40

static __device__ __forceinline__ v16bf frag16(const __bf16* p) {
  v8bf lo = *(const v8bf*)p;
  v8bf hi = *(const v8bf*)(p + 16);  // A: K and K+16 halves
  return __builtin_shufflevector(lo, hi, 0, 1, 2, 3, 4, 5, 6, 7, 8, 9, 10, 11, 12, 13, 14, 15);
}
static __device__ __forceinline__ v16bf frag16c(const __bf16* p) {
  v8bf lo = *(const v8bf*)p;
  v8bf hi = *(const v8bf*)(p + 16);  // B: K and K+16 halves
  return __builtin_shufflevector(lo, hi, 0, 1, 2, 3, 4, 5, 6, 7, 8, 9, 10, 11, 12, 13, 14, 15);
}

// gate/val GEMM + SwiGLU -> h (bf16).  grid (FF/BN, CAP/BM, NE), 512 thr
__global__ void __launch_bounds__(512)
k_gemm1(const __bf16* __restrict__ xe, const float* __restrict__ Wg,
        const float* __restrict__ bg, const float* __restrict__ Wv,
        const float* __restrict__ bv, __bf16* __restrict__ h) {
  __shared__ __bf16 As[2][BM * LDA];
  __shared__ __bf16 Bg[2][BN * LDB];
  __shared__ __bf16 Bv[2][BN * LDB];
  int e = blockIdx.z, mBase = blockIdx.y * BM, nBase = blockIdx.x * BN;
  const __bf16* Aex = xe + ((size_t)e * CAP + mBase) * DM;
  const float*  Wge = Wg + (size_t)e * DM * FF + nBase;
  const float*  Wve = Wv + (size_t)e * DM * FF + nBase;
  int tid = threadIdx.x, lane = tid & 31, wid = tid >> 5;
  int wm = (wid >> 2) * 32, wn = (wid & 3) * 32;
  int half = lane >> 4, mr = lane & 15;
  v8f cg[2][2] = {}, cv[2][2] = {};
  int am = tid >> 2, akc = (tid & 3) * 8;
  int bkk = (tid * 2) >> 5;                 // k-row of this thread's B chunks
  int bn0 = ((tid * 2 + 0) & 31) * 4;
  int bn1 = ((tid * 2 + 1) & 31) * 4;

  // ---- prologue: stage tile 0 into buffer 0
  {
    float4 g0 = *(const float4*)(Wge + (size_t)bkk * FF + bn0);
    float4 g1 = *(const float4*)(Wge + (size_t)bkk * FF + bn1);
    float4 v0 = *(const float4*)(Wve + (size_t)bkk * FF + bn0);
    float4 v1 = *(const float4*)(Wve + (size_t)bkk * FF + bn1);
    async_cp16(Aex + (size_t)am * DM + akc, &As[0][am * LDA + akc]);
    Bg[0][(bn0 + 0) * LDB + bkk] = to_bf16(g0.x);
    Bg[0][(bn0 + 1) * LDB + bkk] = to_bf16(g0.y);
    Bg[0][(bn0 + 2) * LDB + bkk] = to_bf16(g0.z);
    Bg[0][(bn0 + 3) * LDB + bkk] = to_bf16(g0.w);
    Bg[0][(bn1 + 0) * LDB + bkk] = to_bf16(g1.x);
    Bg[0][(bn1 + 1) * LDB + bkk] = to_bf16(g1.y);
    Bg[0][(bn1 + 2) * LDB + bkk] = to_bf16(g1.z);
    Bg[0][(bn1 + 3) * LDB + bkk] = to_bf16(g1.w);
    Bv[0][(bn0 + 0) * LDB + bkk] = to_bf16(v0.x);
    Bv[0][(bn0 + 1) * LDB + bkk] = to_bf16(v0.y);
    Bv[0][(bn0 + 2) * LDB + bkk] = to_bf16(v0.z);
    Bv[0][(bn0 + 3) * LDB + bkk] = to_bf16(v0.w);
    Bv[0][(bn1 + 0) * LDB + bkk] = to_bf16(v1.x);
    Bv[0][(bn1 + 1) * LDB + bkk] = to_bf16(v1.y);
    Bv[0][(bn1 + 2) * LDB + bkk] = to_bf16(v1.z);
    Bv[0][(bn1 + 3) * LDB + bkk] = to_bf16(v1.w);
  }
  async_wait();
  __syncthreads();

  const int STEPS = DM / BK;  // 32
  for (int ks = 0; ks < STEPS; ++ks) {
    int cur = ks & 1, nxt = cur ^ 1;
    int k1 = (ks + 1) * BK;
    bool more = (ks + 1 < STEPS);
    float4 g0, g1, v0, v1;
    if (more) {
      g0 = *(const float4*)(Wge + (size_t)(k1 + bkk) * FF + bn0);
      g1 = *(const float4*)(Wge + (size_t)(k1 + bkk) * FF + bn1);
      v0 = *(const float4*)(Wve + (size_t)(k1 + bkk) * FF + bn0);
      v1 = *(const float4*)(Wve + (size_t)(k1 + bkk) * FF + bn1);
      async_cp16(Aex + (size_t)am * DM + k1 + akc, &As[nxt][am * LDA + akc]);
    }
    // ---- compute from buffer `cur`
    v16bf af[2];
#pragma unroll
    for (int i = 0; i < 2; ++i)
      af[i] = frag16(&As[cur][(wm + i * 16 + mr) * LDA + half * 8]);
#pragma unroll
    for (int j = 0; j < 2; ++j) {
      v16bf bgf = frag16c(&Bg[cur][(wn + j * 16 + mr) * LDB + half * 8]);
      v16bf bvf = frag16c(&Bv[cur][(wn + j * 16 + mr) * LDB + half * 8]);
#pragma unroll
      for (int i = 0; i < 2; ++i) {
        cg[i][j] = __builtin_amdgcn_wmma_f32_16x16x32_bf16(false, af[i], false, bgf,
                                                           (short)0, cg[i][j], false, false);
        cv[i][j] = __builtin_amdgcn_wmma_f32_16x16x32_bf16(false, af[i], false, bvf,
                                                           (short)0, cv[i][j], false, false);
      }
    }
    if (more) {
      Bg[nxt][(bn0 + 0) * LDB + bkk] = to_bf16(g0.x);
      Bg[nxt][(bn0 + 1) * LDB + bkk] = to_bf16(g0.y);
      Bg[nxt][(bn0 + 2) * LDB + bkk] = to_bf16(g0.z);
      Bg[nxt][(bn0 + 3) * LDB + bkk] = to_bf16(g0.w);
      Bg[nxt][(bn1 + 0) * LDB + bkk] = to_bf16(g1.x);
      Bg[nxt][(bn1 + 1) * LDB + bkk] = to_bf16(g1.y);
      Bg[nxt][(bn1 + 2) * LDB + bkk] = to_bf16(g1.z);
      Bg[nxt][(bn1 + 3) * LDB + bkk] = to_bf16(g1.w);
      Bv[nxt][(bn0 + 0) * LDB + bkk] = to_bf16(v0.x);
      Bv[nxt][(bn0 + 1) * LDB + bkk] = to_bf16(v0.y);
      Bv[nxt][(bn0 + 2) * LDB + bkk] = to_bf16(v0.z);
      Bv[nxt][(bn0 + 3) * LDB + bkk] = to_bf16(v0.w);
      Bv[nxt][(bn1 + 0) * LDB + bkk] = to_bf16(v1.x);
      Bv[nxt][(bn1 + 1) * LDB + bkk] = to_bf16(v1.y);
      Bv[nxt][(bn1 + 2) * LDB + bkk] = to_bf16(v1.z);
      Bv[nxt][(bn1 + 3) * LDB + bkk] = to_bf16(v1.w);
    }
    async_wait();
    __syncthreads();
  }
  // ---- epilogue: SwiGLU, store bf16 h
#pragma unroll
  for (int j = 0; j < 2; ++j) {
    int nA = nBase + wn + j * 16 + mr;
    float bgv = bg[(size_t)e * FF + nA];
    float bvv = bv[(size_t)e * FF + nA];
#pragma unroll
    for (int i = 0; i < 2; ++i) {
#pragma unroll
      for (int r = 0; r < 8; ++r) {
        int m = mBase + wm + i * 16 + half * 8 + r;
        float g = cg[i][j][r] + bgv;
        float v = cv[i][j][r] + bvv;
        float hh = v * (g / (1.f + __expf(-g)));
        h[((size_t)e * CAP + m) * FF + nA] = to_bf16(hh);
      }
    }
  }
}

// h @ Wo + bo, scale by prob, scatter-add.  grid (DM/BN, CAP/BM, NE), 512 thr
__global__ void __launch_bounds__(512)
k_gemm2(const __bf16* __restrict__ h, const float* __restrict__ Wo,
        const float* __restrict__ bo, const float* __restrict__ eprob,
        const int* __restrict__ eidx, float* __restrict__ out) {
  __shared__ __bf16 As[2][BM * LDA];
  __shared__ __bf16 Bt[2][BN * LDB];
  int e = blockIdx.z, mBase = blockIdx.y * BM, nBase = blockIdx.x * BN;
  const __bf16* Ah  = h + ((size_t)e * CAP + mBase) * FF;
  const float*  Woe = Wo + (size_t)e * FF * DM + nBase;
  int tid = threadIdx.x, lane = tid & 31, wid = tid >> 5;
  int wm = (wid >> 2) * 32, wn = (wid & 3) * 32;
  int half = lane >> 4, mr = lane & 15;
  v8f cc[2][2] = {};
  int am = tid >> 2, akc = (tid & 3) * 8;
  int bkk = (tid * 2) >> 5;
  int bn0 = ((tid * 2 + 0) & 31) * 4;
  int bn1 = ((tid * 2 + 1) & 31) * 4;

  {
    float4 w0 = *(const float4*)(Woe + (size_t)bkk * DM + bn0);
    float4 w1 = *(const float4*)(Woe + (size_t)bkk * DM + bn1);
    async_cp16(Ah + (size_t)am * FF + akc, &As[0][am * LDA + akc]);
    Bt[0][(bn0 + 0) * LDB + bkk] = to_bf16(w0.x);
    Bt[0][(bn0 + 1) * LDB + bkk] = to_bf16(w0.y);
    Bt[0][(bn0 + 2) * LDB + bkk] = to_bf16(w0.z);
    Bt[0][(bn0 + 3) * LDB + bkk] = to_bf16(w0.w);
    Bt[0][(bn1 + 0) * LDB + bkk] = to_bf16(w1.x);
    Bt[0][(bn1 + 1) * LDB + bkk] = to_bf16(w1.y);
    Bt[0][(bn1 + 2) * LDB + bkk] = to_bf16(w1.z);
    Bt[0][(bn1 + 3) * LDB + bkk] = to_bf16(w1.w);
  }
  async_wait();
  __syncthreads();

  const int STEPS = FF / BK;  // 128
  for (int ks = 0; ks < STEPS; ++ks) {
    int cur = ks & 1, nxt = cur ^ 1;
    int k1 = (ks + 1) * BK;
    bool more = (ks + 1 < STEPS);
    float4 w0, w1;
    if (more) {
      w0 = *(const float4*)(Woe + (size_t)(k1 + bkk) * DM + bn0);
      w1 = *(const float4*)(Woe + (size_t)(k1 + bkk) * DM + bn1);
      async_cp16(Ah + (size_t)am * FF + k1 + akc, &As[nxt][am * LDA + akc]);
    }
    v16bf af[2];
#pragma unroll
    for (int i = 0; i < 2; ++i)
      af[i] = frag16(&As[cur][(wm + i * 16 + mr) * LDA + half * 8]);
#pragma unroll
    for (int j = 0; j < 2; ++j) {
      v16bf bf = frag16c(&Bt[cur][(wn + j * 16 + mr) * LDB + half * 8]);
#pragma unroll
      for (int i = 0; i < 2; ++i)
        cc[i][j] = __builtin_amdgcn_wmma_f32_16x16x32_bf16(false, af[i], false, bf,
                                                           (short)0, cc[i][j], false, false);
    }
    if (more) {
      Bt[nxt][(bn0 + 0) * LDB + bkk] = to_bf16(w0.x);
      Bt[nxt][(bn0 + 1) * LDB + bkk] = to_bf16(w0.y);
      Bt[nxt][(bn0 + 2) * LDB + bkk] = to_bf16(w0.z);
      Bt[nxt][(bn0 + 3) * LDB + bkk] = to_bf16(w0.w);
      Bt[nxt][(bn1 + 0) * LDB + bkk] = to_bf16(w1.x);
      Bt[nxt][(bn1 + 1) * LDB + bkk] = to_bf16(w1.y);
      Bt[nxt][(bn1 + 2) * LDB + bkk] = to_bf16(w1.z);
      Bt[nxt][(bn1 + 3) * LDB + bkk] = to_bf16(w1.w);
    }
    async_wait();
    __syncthreads();
  }
#pragma unroll
  for (int j = 0; j < 2; ++j) {
    int nA = nBase + wn + j * 16 + mr;
    float bov = bo[(size_t)e * DM + nA];
#pragma unroll
    for (int i = 0; i < 2; ++i) {
#pragma unroll
      for (int r = 0; r < 8; ++r) {
        int m = mBase + wm + i * 16 + half * 8 + r;
        int tok = eidx[e * CAP + m];
        if (tok >= 0) {
          float pv = eprob[e * CAP + m];
          float o = (cc[i][j][r] + bov) * pv;
          atomicAdd(out + (size_t)tok * DM + nA, o);
        }
      }
    }
  }
}

// ---------------------------------------------------------------- launch
extern "C" void kernel_launch(void* const* d_in, const int* in_sizes, int n_in,
                              void* d_out, int out_size, void* d_ws, size_t ws_size,
                              hipStream_t stream) {
  (void)in_sizes; (void)n_in; (void)out_size; (void)ws_size;
  const float* x  = (const float*)d_in[0];
  const float* Wr = (const float*)d_in[1];
  const float* br = (const float*)d_in[2];
  const float* Wg = (const float*)d_in[3];
  const float* bg = (const float*)d_in[4];
  const float* Wv = (const float*)d_in[5];
  const float* bv = (const float*)d_in[6];
  const float* Wo = (const float*)d_in[7];
  const float* bo = (const float*)d_in[8];

  // d_out tuple layout: out | logits | probs | expert_probs | expert_indices
  float* out     = (float*)d_out;
  float* logits  = out + (size_t)NTOK * DM;
  float* probs   = logits + (size_t)NTOK * NE;
  float* eprob_o = probs + (size_t)NTOK * NE;
  int*   eidx_o  = (int*)(eprob_o + NE * CAP);

  char* ws = (char*)d_ws;
  int*    tki   = (int*)(ws);
  float*  tkp   = (float*)(ws + 32768);
  int*    eidx  = (int*)(ws + 65536);
  float*  eprob = (float*)(ws + 106496);
  __bf16* xe    = (__bf16*)(ws + 147456);
  __bf16* hb    = (__bf16*)(ws + 147456 + (size_t)NE * CAP * DM * 2);

  k_zero<<<(NTOK * DM / 4 + 255) / 256, 256, 0, stream>>>(out, NTOK * DM / 4);
  k_router<<<NTOK / 256, 256, 0, stream>>>(x, Wr, br, logits, probs, tki, tkp);
  k_dispatch<<<1, DTH, 0, stream>>>(tki, tkp, eidx_o, eprob_o, eidx, eprob);
  k_gather<<<NE * CAP, 256, 0, stream>>>(x, eidx, xe);
  k_gemm1<<<dim3(FF / BN, CAP / BM, NE), 512, 0, stream>>>(xe, Wg, bg, Wv, bv, hb);
  k_gemm2<<<dim3(DM / BN, CAP / BM, NE), 512, 0, stream>>>(hb, Wo, bo, eprob, eidx, out);
}
